// ReprogrammingLayer_55997783605916
// MI455X (gfx1250) — matrix operand order repaired and checked by reading, not compiled, not run on hardware
//
#include <hip/hip_runtime.h>
#include <hip/hip_bf16.h>

typedef __bf16 bf16;
typedef __attribute__((ext_vector_type(16))) __bf16 v16bf;
typedef __attribute__((ext_vector_type(8)))  float  v8f;

union FragU { v16bf v; uint4 q[2]; };

// Low 32 bits of a generic pointer to LDS = DS byte offset (flat-LDS aperture rule).
static __device__ inline unsigned lds_off(const void* p) { return (unsigned)(size_t)p; }

// CDNA5 async copy: global -> LDS, 16B per lane, tracked by ASYNCcnt.
static __device__ inline void async_b128(unsigned lds_dst, const void* gsrc) {
  asm volatile("global_load_async_to_lds_b128 %0, %1, off"
               :: "v"(lds_dst), "v"((unsigned long long)(size_t)gsrc)
               : "memory");
}
#define WAIT_ASYNCCNT0() asm volatile("s_wait_asynccnt 0x0" ::: "memory")
// Wave-local LDS ordering fence (DS ops are in-order per wave; this pins the compiler too).
#define LDS_FENCE() asm volatile("s_wait_dscnt 0x0" ::: "memory")

// A-fragment (16x32 bf16, MxK): lane L -> row m = base + (L&15);
// VGPR0-3 hold K = kh..kh+7, VGPR4-7 hold K = kh+16..kh+23, kh = (L<16)?0:8.
static __device__ inline v16bf lds_a_frag(const bf16* p0, int ld, int m, int kbase) {
  const int lane = threadIdx.x & 31;
  const bf16* p = p0 + (m + (lane & 15)) * ld + kbase + ((lane < 16) ? 0 : 8);
  FragU f;
  f.q[0] = *(const uint4*)(p);
  f.q[1] = *(const uint4*)(p + 16);
  return f.v;
}

// B-fragment (32x16 bf16, KxN), LDS stored [n][k]: lane L -> col n = base + (L&15);
// VGPR0-3 hold K = kh..kh+7, VGPR4-7 hold K = kh+16..kh+23, kh = (L<16)?0:8.
static __device__ inline v16bf lds_b_frag(const bf16* p0, int ld, int n, int kbase) {
  const int lane = threadIdx.x & 31;
  const bf16* p = p0 + (n + (lane & 15)) * ld + kbase + ((lane < 16) ? 0 : 8);
  FragU f;
  f.q[0] = *(const uint4*)(p);
  f.q[1] = *(const uint4*)(p + 16);
  return f.v;
}

#define WMMA_BF16(a, b, c) \
  __builtin_amdgcn_wmma_f32_16x16x32_bf16(false, (a), false, (b), (short)0, (c), false, false)

// ---------------------------------------------------------------------------
// Tiled GEMM: C[M,N] = (A[M,K] @ B[K,N] + bias[N]) * out_scale
// A is f32 (converted to bf16 through VGPRs) or bf16 (async-copied straight
// to LDS, double-buffered). B/bias f32. Block tile 128x128, K-step 32,
// 8 waves each computing a 32x64 sub-tile via v_wmma_f32_16x16x32_bf16.
// ---------------------------------------------------------------------------
#define GLDA 40  // LDS row stride in bf16 (32 data + 8 pad), 80B rows

template <bool OUT_BF16, bool A_BF16>
__global__ __launch_bounds__(256) void gemm128(const void* __restrict__ Ain,
                                               const float* __restrict__ B,
                                               const float* __restrict__ bias,
                                               void* __restrict__ Cout,
                                               int M, int N, int K, float out_scale) {
  constexpr int ABUFS = A_BF16 ? 2 : 1;
  __shared__ bf16 As[ABUFS * 128 * GLDA];  // [buf][m][k]
  __shared__ bf16 Bs[128 * GLDA];          // [n][k] (transposed on store)

  const float* Af = (const float*)Ain;
  const bf16*  Ah = (const bf16*)Ain;

  const int tid  = threadIdx.x;
  const int lane = tid & 31;
  const int wave = tid >> 5;
  const int m0 = blockIdx.y * 128;
  const int n0 = blockIdx.x * 128;
  const int wm = (wave & 3) * 32;   // 4 waves along M
  const int wn = (wave >> 2) * 64;  // 2 waves along N

  const v8f vzero = {0.f, 0.f, 0.f, 0.f, 0.f, 0.f, 0.f, 0.f};
  v8f acc[2][4];
#pragma unroll
  for (int s = 0; s < 2; ++s)
#pragma unroll
    for (int t = 0; t < 4; ++t) acc[s][t] = vzero;

  float4 ra[4], rb[4];

  auto loadAf = [&](int kt) {  // f32 A -> regs
#pragma unroll
    for (int i = 0; i < 4; ++i) {
      int g = tid + i * 256;
      int row = g >> 3, c4 = (g & 7) * 4;
      int m = m0 + row;
      ra[i] = (m < M) ? *(const float4*)(Af + (size_t)m * K + kt * 32 + c4)
                      : make_float4(0.f, 0.f, 0.f, 0.f);
    }
  };
  auto stageAf = [&]() {  // regs -> bf16 LDS
#pragma unroll
    for (int i = 0; i < 4; ++i) {
      int g = tid + i * 256;
      int row = g >> 3, c4 = (g & 7) * 4;
      bf16 h[4] = {(bf16)ra[i].x, (bf16)ra[i].y, (bf16)ra[i].z, (bf16)ra[i].w};
      *(uint2*)(&As[row * GLDA + c4]) = *(const uint2*)h;
    }
  };
  auto issueAh = [&](int kt, int buf) {  // bf16 A: async copy straight to LDS
#pragma unroll
    for (int i = 0; i < 2; ++i) {
      int g = tid + i * 256;
      int row = g >> 2, c8 = (g & 3) * 8;
      int m = m0 + row;
      if (m >= M) m = M - 1;  // clamp; stores are guarded in epilogue
      async_b128(lds_off(&As[buf * 128 * GLDA + row * GLDA + c8]),
                 Ah + (size_t)m * K + kt * 32 + c8);
    }
  };
  auto loadB = [&](int kt) {
#pragma unroll
    for (int i = 0; i < 4; ++i) {
      int g = tid + i * 256;
      int kk = g >> 5, c4 = (g & 31) * 4;
      rb[i] = *(const float4*)(B + (size_t)(kt * 32 + kk) * N + n0 + c4);
    }
  };
  auto stageB = [&]() {
#pragma unroll
    for (int i = 0; i < 4; ++i) {
      int g = tid + i * 256;
      int kk = g >> 5, c4 = (g & 31) * 4;
      Bs[(c4 + 0) * GLDA + kk] = (bf16)rb[i].x;
      Bs[(c4 + 1) * GLDA + kk] = (bf16)rb[i].y;
      Bs[(c4 + 2) * GLDA + kk] = (bf16)rb[i].z;
      Bs[(c4 + 3) * GLDA + kk] = (bf16)rb[i].w;
    }
  };

  const int nk = K >> 5;
  if (A_BF16) issueAh(0, 0); else loadAf(0);
  loadB(0);

  for (int kt = 0; kt < nk; ++kt) {
    const int cur = A_BF16 ? (kt & 1) : 0;
    if (!A_BF16) stageAf();
    stageB();
    if (kt + 1 < nk) {
      if (!A_BF16) loadAf(kt + 1);
      loadB(kt + 1);
    }
    if (A_BF16) WAIT_ASYNCCNT0();  // A tile landed in LDS before barrier signal
    __syncthreads();
    if (A_BF16 && kt + 1 < nk) issueAh(kt + 1, cur ^ 1);  // safe: buf^1 idle since kt-1's end barrier

    const bf16* Ab = &As[cur * 128 * GLDA];
    v16bf af0 = lds_a_frag(Ab, GLDA, wm, 0);
    v16bf af1 = lds_a_frag(Ab, GLDA, wm + 16, 0);
    v16bf bfr[4];
#pragma unroll
    for (int t = 0; t < 4; ++t) bfr[t] = lds_b_frag(Bs, GLDA, wn + t * 16, 0);
#pragma unroll
    for (int t = 0; t < 4; ++t) {
      acc[0][t] = WMMA_BF16(af0, bfr[t], acc[0][t]);
      acc[1][t] = WMMA_BF16(af1, bfr[t], acc[1][t]);
    }
    __syncthreads();
  }

  // Epilogue: C layout -> VGPR r: row = r + 8*(lane>=16), col = lane&15 per 16-tile.
  const int rowoff = (lane < 16) ? 0 : 8;
#pragma unroll
  for (int s = 0; s < 2; ++s) {
#pragma unroll
    for (int t = 0; t < 4; ++t) {
      int n = n0 + wn + t * 16 + (lane & 15);
      float bb = bias[n];
#pragma unroll
      for (int r = 0; r < 8; ++r) {
        int m = m0 + wm + s * 16 + r + rowoff;
        if (m < M) {
          float valf = (acc[s][t][r] + bb) * out_scale;
          if (OUT_BF16)
            ((bf16*)Cout)[(size_t)m * N + n] = (bf16)valf;
          else
            ((float*)Cout)[(size_t)m * N + n] = valf;
        }
      }
    }
  }
}

// ---------------------------------------------------------------------------
// Flash attention per (b,h) x 128 L-rows. q pre-scaled by 1/sqrt(E).
// Q + K staged via async-LDS copies (K double-buffered); V double-buffered
// through registers with a transpose store; one __syncthreads per S-chunk.
// Q's LDS region is reused as the per-wave P strips after the prologue.
// Buffer pointers are computed arithmetically from the ping-pong index
// (no LDS-pointer aggregates -> no static addrspacecast initializers).
// ---------------------------------------------------------------------------
__global__ __launch_bounds__(256) void attn128(const bf16* __restrict__ q,
                                               const bf16* __restrict__ kmat,
                                               const bf16* __restrict__ vmat,
                                               bf16* __restrict__ o, int S) {
  constexpr int H = 12, E = 64, L = 1024, DI = 768;
  constexpr int LDK = 72;
  constexpr int CHUNK = 64 * LDK;  // elements per K/V chunk buffer
  // [Ks0 | Ks1 | Vs0 | Vs1 | Qs/Ps(128x72)] = 55296 B
  __shared__ __align__(16) bf16 smem[4 * CHUNK + 128 * LDK];

  const int tid = threadIdx.x, lane = tid & 31, wave = tid >> 5;
  const int b = blockIdx.x / H, h = blockIdx.x % H;
  const int l0 = blockIdx.y * 128;
  const size_t qbase = ((size_t)b * L + l0) * DI + h * E;
  const size_t kvcol = (size_t)h * E;
  const int Sm1 = S - 1;

  bf16* const Qs = &smem[4 * CHUNK];  // 128x72; becomes P strips after prologue

  // Async-load Q tile (128x64 bf16) straight into LDS.
#pragma unroll
  for (int i = 0; i < 4; ++i) {
    int g = tid + i * 256;
    int row = g >> 3, c8 = (g & 7) * 8;
    async_b128(lds_off(Qs + row * LDK + c8), q + qbase + (size_t)row * DI + c8);
  }

  auto issueK = [&](int sc, int buf) {
    bf16* dst = &smem[buf * CHUNK];
#pragma unroll
    for (int i = 0; i < 2; ++i) {
      int g = tid + i * 256;
      int row = g >> 3, c8 = (g & 7) * 8;
      int s = sc * 64 + row;
      if (s > Sm1) s = Sm1;  // clamp; padded rows masked to -inf after WMMA
      async_b128(lds_off(dst + row * LDK + c8), kmat + (size_t)s * DI + kvcol + c8);
    }
  };
  uint4 vreg[2];
  auto loadV = [&](int sc) {
#pragma unroll
    for (int i = 0; i < 2; ++i) {
      int g = tid + i * 256;
      int row = g >> 3, c8 = (g & 7) * 8;
      int s = sc * 64 + row;
      if (s > Sm1) s = Sm1;  // clamp; P==0 for padded rows
      vreg[i] = *(const uint4*)(vmat + (size_t)s * DI + kvcol + c8);
    }
  };
  auto storeV = [&](int buf) {  // transpose to [e][s]
    bf16* dst = &smem[(2 + buf) * CHUNK];
#pragma unroll
    for (int i = 0; i < 2; ++i) {
      int g = tid + i * 256;
      int row = g >> 3, c8 = (g & 7) * 8;
      const bf16* hp = (const bf16*)&vreg[i];
#pragma unroll
      for (int j = 0; j < 8; ++j) dst[(c8 + j) * LDK + row] = hp[j];
    }
  };

  issueK(0, 0);
  loadV(0);
  WAIT_ASYNCCNT0();  // Q + K0 resident
  __syncthreads();

  // Q fragments live in registers for the whole S loop.
  v16bf qa0 = lds_a_frag(Qs, LDK, wave * 16, 0);
  v16bf qa1 = lds_a_frag(Qs, LDK, wave * 16, 32);

  const v8f vzero = {0.f, 0.f, 0.f, 0.f, 0.f, 0.f, 0.f, 0.f};
  float mprev[8], lsum[8];
#pragma unroll
  for (int r = 0; r < 8; ++r) { mprev[r] = -1e30f; lsum[r] = 0.f; }
  v8f oacc[4] = {vzero, vzero, vzero, vzero};
  const int rowoff = (lane < 16) ? 0 : 8;
  bf16* const Pw = Qs + wave * 16 * LDK;  // this wave's 16x64 P strip

  const int nchunk = (S + 63) >> 6;
  for (int sc = 0; sc < nchunk; ++sc) {
    const int cur = sc & 1;
    const int s0 = sc * 64;
    storeV(cur);           // Vs[cur] idle since chunk sc-2 (barrier at sc-1 separates)
    WAIT_ASYNCCNT0();      // Ks[cur] (issued last iteration) resident
    __syncthreads();
    if (sc + 1 < nchunk) {
      issueK(sc + 1, cur ^ 1);  // buf^1 idle: its readers finished before this barrier
      loadV(sc + 1);
    }

    const bf16* Ks = &smem[cur * CHUNK];
    const bf16* Vs = &smem[(2 + cur) * CHUNK];

    // scores [16 x 64] for this wave's strip
    v8f sct[4] = {vzero, vzero, vzero, vzero};
#pragma unroll
    for (int t = 0; t < 4; ++t) {
      v16bf kb0 = lds_b_frag(Ks, LDK, t * 16, 0);
      v16bf kb1 = lds_b_frag(Ks, LDK, t * 16, 32);
      sct[t] = WMMA_BF16(qa0, kb0, sct[t]);
      sct[t] = WMMA_BF16(qa1, kb1, sct[t]);
    }

    // mask + row max (xor-shuffles stay within 16-lane halves = same rows)
    float rmax[8];
#pragma unroll
    for (int r = 0; r < 8; ++r) rmax[r] = -1e30f;
#pragma unroll
    for (int t = 0; t < 4; ++t) {
      int n = s0 + t * 16 + (lane & 15);
#pragma unroll
      for (int r = 0; r < 8; ++r) {
        float x = sct[t][r];
        if (n >= S) { x = -1e30f; sct[t][r] = x; }
        rmax[r] = fmaxf(rmax[r], x);
      }
    }
#pragma unroll
    for (int off = 1; off < 16; off <<= 1)
#pragma unroll
      for (int r = 0; r < 8; ++r) rmax[r] = fmaxf(rmax[r], __shfl_xor(rmax[r], off, 32));

    float corr[8], rsum[8];
#pragma unroll
    for (int r = 0; r < 8; ++r) {
      float mn = fmaxf(mprev[r], rmax[r]);
      corr[r] = __expf(mprev[r] - mn);
      mprev[r] = mn;
      rsum[r] = 0.f;
    }

    // P = exp(scores - m) -> per-wave LDS strip in A-source layout [m][s]
#pragma unroll
    for (int t = 0; t < 4; ++t) {
#pragma unroll
      for (int r = 0; r < 8; ++r) {
        float p = __expf(sct[t][r] - mprev[r]);
        rsum[r] += p;
        Pw[(r + rowoff) * LDK + t * 16 + (lane & 15)] = (bf16)p;
      }
    }
#pragma unroll
    for (int off = 1; off < 16; off <<= 1)
#pragma unroll
      for (int r = 0; r < 8; ++r) rsum[r] += __shfl_xor(rsum[r], off, 32);
#pragma unroll
    for (int r = 0; r < 8; ++r) lsum[r] = lsum[r] * corr[r] + rsum[r];
#pragma unroll
    for (int te = 0; te < 4; ++te)
#pragma unroll
      for (int r = 0; r < 8; ++r) oacc[te][r] *= corr[r];

    LDS_FENCE();  // P strip is wave-private; DS in-order per wave + pin compiler

    // O += P @ V
    v16bf pa0 = lds_a_frag(Pw, LDK, 0, 0);
    v16bf pa1 = lds_a_frag(Pw, LDK, 0, 32);
#pragma unroll
    for (int te = 0; te < 4; ++te) {
      v16bf vb0 = lds_b_frag(Vs, LDK, te * 16, 0);
      v16bf vb1 = lds_b_frag(Vs, LDK, te * 16, 32);
      oacc[te] = WMMA_BF16(pa0, vb0, oacc[te]);
      oacc[te] = WMMA_BF16(pa1, vb1, oacc[te]);
    }
  }

  // Normalize and store bf16 [B,L,H,E]
#pragma unroll
  for (int te = 0; te < 4; ++te) {
    int e = te * 16 + (lane & 15);
#pragma unroll
    for (int r = 0; r < 8; ++r) {
      int l = l0 + wave * 16 + r + rowoff;
      o[((size_t)b * L + l) * DI + kvcol + e] = (bf16)(oacc[te][r] / lsum[r]);
    }
  }
}

// ---------------------------------------------------------------------------
extern "C" void kernel_launch(void* const* d_in, const int* in_sizes, int n_in,
                              void* d_out, int out_size, void* d_ws, size_t ws_size,
                              hipStream_t stream) {
  (void)in_sizes; (void)n_in; (void)out_size; (void)ws_size;
  const float* tgt = (const float*)d_in[0];   // [8,1024,768]
  const float* src = (const float*)d_in[1];   // [1000,4096]
  const float* val = (const float*)d_in[2];   // [1000,4096]
  const float* Wq  = (const float*)d_in[3];
  const float* bq  = (const float*)d_in[4];
  const float* Wk  = (const float*)d_in[5];
  const float* bk  = (const float*)d_in[6];
  const float* Wv  = (const float*)d_in[7];
  const float* bv  = (const float*)d_in[8];
  const float* Wo  = (const float*)d_in[9];
  const float* bo  = (const float*)d_in[10];
  float* out = (float*)d_out;                 // [8,1024,4096]

  char* ws = (char*)d_ws;
  size_t off = 0;
  auto alloc = [&](size_t bytes) {
    size_t o = off;
    off = (off + bytes + 255) & ~(size_t)255;
    return o;
  };
  bf16* qws = (bf16*)(ws + alloc((size_t)8192 * 768 * 2));  // q bf16 [8192,768]
  bf16* kws = (bf16*)(ws + alloc((size_t)1000 * 768 * 2));  // k bf16 [1000,768]
  bf16* vws = (bf16*)(ws + alloc((size_t)1000 * 768 * 2));  // v bf16 [1000,768]
  bf16* aws = (bf16*)(ws + alloc((size_t)8192 * 768 * 2));  // attn out bf16 [8192,768]

  const dim3 blk(256);
  const float qscale = 0.125f;  // 1/sqrt(64), folded into Q projection

  // Q = (target @ Wq + bq) * scale  -> bf16
  gemm128<true, false><<<dim3(768 / 128, 8192 / 128), blk, 0, stream>>>(
      tgt, Wq, bq, qws, 8192, 768, 768, qscale);
  // K = source @ Wk + bk -> bf16
  gemm128<true, false><<<dim3(768 / 128, (1000 + 127) / 128), blk, 0, stream>>>(
      src, Wk, bk, kws, 1000, 768, 4096, 1.0f);
  // V = value @ Wv + bv -> bf16
  gemm128<true, false><<<dim3(768 / 128, (1000 + 127) / 128), blk, 0, stream>>>(
      val, Wv, bv, vws, 1000, 768, 4096, 1.0f);
  // Flash attention -> bf16 [8192,768]
  attn128<<<dim3(8 * 12, 1024 / 128), blk, 0, stream>>>(qws, kws, vws, aws, 1000);
  // out = attn @ Wo + bo -> f32 (A staged via async-LDS bf16 path)
  gemm128<false, true><<<dim3(4096 / 128, 8192 / 128), blk, 0, stream>>>(
      aws, Wo, bo, out, 8192, 4096, 768, 1.0f);
}
